// A_GCN_85203561218338
// MI455X (gfx1250) — matrix operand order repaired and checked by reading, not compiled, not run on hardware
//
#include <hip/hip_runtime.h>
#include <hip/hip_bf16.h>
#include <math.h>

// ---------------------------------------------------------------------------
// Adaptive GCN (AGCRN-style) on MI455X / gfx1250.
// All GEMM-shaped work runs through v_wmma_f32_16x16x32_f16 (wave32).
// ---------------------------------------------------------------------------

typedef __attribute__((ext_vector_type(16))) _Float16 v16h;
typedef __attribute__((ext_vector_type(8)))  _Float16 v8h;
typedef __attribute__((ext_vector_type(8)))  float    v8f;

#define BB    64
#define NN    400
#define DIN_  64
#define DOUT_ 64
#define DP_   16
#define DU_   16
#define DF_   16
#define NP    416            // N padded to 13*32 for f16 WMMA K-chunks
#define KI_   192            // 3*64 flattened (k,i)
#define MT    25             // 16-node tiles per batch (400/16)

// workspace layout (bytes; every region a multiple of 256)
#define OFF_FE32  ((size_t)0)
#define SZ_FE32   ((size_t)BB*NN*DF_*4)
#define OFF_FE16  (OFF_FE32 + SZ_FE32)
#define SZ_FE16   ((size_t)BB*NN*DF_*2)
#define OFF_SUP   (OFF_FE16 + SZ_FE16)
#define SZ_SUP    ((size_t)BB*NN*NP*2)
#define OFF_XT    (OFF_SUP + SZ_SUP)
#define SZ_XT     ((size_t)BB*DIN_*NP*2)
#define OFF_T1XT  (OFF_XT + SZ_XT)
#define SZ_T1XT   ((size_t)BB*DIN_*NP*2)
#define OFF_XG    (OFF_T1XT + SZ_T1XT)
#define SZ_XG     ((size_t)BB*NN*KI_*2)
#define OFF_WPT   (OFF_XG + SZ_XG)
#define SZ_WPT    ((size_t)DF_*DOUT_*KI_*2)

__device__ __forceinline__ v16h pack16(v8h lo, v8h hi) {
  v16h r;
#pragma unroll
  for (int e = 0; e < 8; e++) { r[e] = lo[e]; r[e + 8] = hi[e]; }
  return r;
}

// ---------------------------------------------------------------------------
// fe[b,n,f] = sum_d embed[b,n,d] * W_fe[f,d] + b_fe[f]
// ---------------------------------------------------------------------------
__global__ void k_fe(const float* __restrict__ pre, const float* __restrict__ unk,
                     const float* __restrict__ Wfe, const float* __restrict__ bfe,
                     float* __restrict__ fe32, _Float16* __restrict__ fe16) {
  int t = blockIdx.x * blockDim.x + threadIdx.x;
  if (t >= BB * NN * DF_) return;
  int f = t % DF_;
  int n = (t / DF_) % NN;
  int b = t / (DF_ * NN);
  const float* e0 = pre + ((size_t)b * NN + n) * DP_;
  const float* e1 = unk + (size_t)n * DU_;
  const float* w  = Wfe + (size_t)f * (DP_ + DU_);
  float acc = bfe[f];
#pragma unroll
  for (int d = 0; d < DP_; d++) acc += e0[d] * w[d];
#pragma unroll
  for (int d = 0; d < DU_; d++) acc += e1[d] * w[DP_ + d];
  fe32[t] = acc;
  fe16[t] = (_Float16)acc;
}

// ---------------------------------------------------------------------------
// xT_f16[b,i,m] (m padded to 416 with zeros), zero t1xT pad rows,
// x_g[b,n,0,i] = x (Chebyshev T0).
// ---------------------------------------------------------------------------
__global__ void k_xprep(const float* __restrict__ x, _Float16* __restrict__ xT,
                        _Float16* __restrict__ t1xT, _Float16* __restrict__ xg) {
  int t = blockIdx.x * blockDim.x + threadIdx.x;
  if (t >= BB * DIN_ * NP) return;
  int m = t % NP;
  int i = (t / NP) % DIN_;
  int b = t / (NP * DIN_);
  float v = 0.0f;
  if (m < NN) {
    v = x[((size_t)b * NN + m) * DIN_ + i];
    xg[((size_t)b * NN + m) * KI_ + i] = (_Float16)v;   // k = 0 slot
  } else {
    t1xT[t] = (_Float16)0.0f;                           // pad rows stay zero
  }
  xT[t] = (_Float16)v;
}

// ---------------------------------------------------------------------------
// WpT[f,o,ki] = weights_pool[f,k,i,o] as f16 (contiguous ki for B-operands)
// ---------------------------------------------------------------------------
__global__ void k_wpt(const float* __restrict__ wp, _Float16* __restrict__ wpt) {
  int t = blockIdx.x * blockDim.x + threadIdx.x;
  if (t >= DF_ * DOUT_ * KI_) return;
  int ki = t % KI_;
  int o  = (t / KI_) % DOUT_;
  int f  = t / (KI_ * DOUT_);
  int k = ki / DIN_, i = ki % DIN_;
  wpt[t] = (_Float16)wp[(((size_t)f * 3 + k) * DIN_ + i) * DOUT_ + o];
}

// ---------------------------------------------------------------------------
// supports = softmax(relu(fe @ fe^T)) ; one wave per (b, 16-row stripe).
// WMMA K = 16 (zero-padded to 32). Output f16, cols 400..415 zeroed.
// ---------------------------------------------------------------------------
__global__ void __launch_bounds__(32) k_adj(const _Float16* __restrict__ fe16,
                                            _Float16* __restrict__ sup) {
  __shared__ float srow[16 * NP];
  int blk = blockIdx.x;
  int b = blk / MT, mt = blk % MT;
  int L = threadIdx.x, lm = L & 15, half = L >> 4;
  int row = mt * 16 + lm;

  // A operand: lane L holds M=lm, K = half*8 + e (+8 for e>=8); upper K zero.
  v16h a;
  {
    v8h lo = *(const v8h*)(fe16 + ((size_t)b * NN + row) * DF_ + half * 8);
#pragma unroll
    for (int e = 0; e < 8; e++) { a[e] = lo[e]; a[e + 8] = (_Float16)0.0f; }
  }

  for (int ct = 0; ct < MT; ct++) {
    int col = ct * 16 + lm;
    v16h bop;
    {   // K = half*8 + e for e<8 (fe of the column node); upper K zero
      v8h c0 = *(const v8h*)(fe16 + ((size_t)b * NN + col) * DF_ + half * 8);
#pragma unroll
      for (int e = 0; e < 8; e++) { bop[e] = c0[e]; bop[e + 8] = (_Float16)0.0f; }
    }
    v8f c = {};
    c = __builtin_amdgcn_wmma_f32_16x16x32_f16(false, a, false, bop,
                                               (short)0, c, false, false);
#pragma unroll
    for (int j = 0; j < 8; j++) {
      int r = j + half * 8;
      srow[r * NP + ct * 16 + lm] = fmaxf(c[j], 0.0f);   // relu
    }
  }
  __syncthreads();

  // row softmax: two lanes per row (interleaved columns), combine via shfl_xor 16
  int r = lm;
  float mx = 0.0f;                      // relu'd values are >= 0
  for (int c = half; c < NN; c += 2) mx = fmaxf(mx, srow[r * NP + c]);
  mx = fmaxf(mx, __shfl_xor(mx, 16));
  float sm = 0.0f;
  for (int c = half; c < NN; c += 2) sm += expf(srow[r * NP + c] - mx);
  sm += __shfl_xor(sm, 16);
  float inv = 1.0f / sm;
  size_t base = ((size_t)b * NN + mt * 16 + r) * NP;
  for (int c = half; c < NN; c += 2)
    sup[base + c] = (_Float16)(expf(srow[r * NP + c] - mx) * inv);
  for (int c = NN + half; c < NP; c += 2)
    sup[base + c] = (_Float16)0.0f;     // K padding for downstream GEMMs
}

// ---------------------------------------------------------------------------
// t1x = supports @ x. One wave per (b, 16-row stripe), 4 N-tiles x 13 K-chunks.
// Writes x_g slot 1 and transposed t1x (B-operand for the next GEMM).
// ---------------------------------------------------------------------------
__global__ void __launch_bounds__(32) k_t1x(const _Float16* __restrict__ sup,
                                            const _Float16* __restrict__ xT,
                                            _Float16* __restrict__ xg,
                                            _Float16* __restrict__ t1xT) {
  int blk = blockIdx.x;
  int b = blk / MT, mt = blk % MT;
  int L = threadIdx.x, lm = L & 15, half = L >> 4;
  int row = mt * 16 + lm;
  const _Float16* srowp = sup + ((size_t)b * NN + row) * NP;

  v8f acc[4] = {};
  for (int kt = 0; kt < 13; kt++) {
    int kb = kt * 32;
    v8h a0 = *(const v8h*)(srowp + kb + half * 16);
    v8h a1 = *(const v8h*)(srowp + kb + half * 16 + 8);
    v16h a = pack16(a0, a1);
#pragma unroll
    for (int nt = 0; nt < 4; nt++) {
      const _Float16* bp = xT + ((size_t)b * DIN_ + nt * 16 + lm) * NP + kb + half * 16;
      v16h bop = pack16(*(const v8h*)bp, *(const v8h*)(bp + 8));
      acc[nt] = __builtin_amdgcn_wmma_f32_16x16x32_f16(false, a, false, bop,
                                                       (short)0, acc[nt], false, false);
    }
  }
#pragma unroll
  for (int nt = 0; nt < 4; nt++) {
    int i = nt * 16 + lm;
#pragma unroll
    for (int j = 0; j < 8; j++) {
      int node = mt * 16 + j + half * 8;
      _Float16 v = (_Float16)acc[nt][j];
      xg[((size_t)b * NN + node) * KI_ + 64 + i] = v;          // k = 1 slot
      t1xT[((size_t)b * DIN_ + i) * NP + node]  = v;
    }
  }
}

// ---------------------------------------------------------------------------
// t2x = 2 * supports @ t1x - x  -> x_g slot 2.
// ---------------------------------------------------------------------------
__global__ void __launch_bounds__(32) k_t2x(const _Float16* __restrict__ sup,
                                            const _Float16* __restrict__ t1xT,
                                            const float* __restrict__ x,
                                            _Float16* __restrict__ xg) {
  int blk = blockIdx.x;
  int b = blk / MT, mt = blk % MT;
  int L = threadIdx.x, lm = L & 15, half = L >> 4;
  int row = mt * 16 + lm;
  const _Float16* srowp = sup + ((size_t)b * NN + row) * NP;

  v8f acc[4] = {};
  for (int kt = 0; kt < 13; kt++) {
    int kb = kt * 32;
    v8h a0 = *(const v8h*)(srowp + kb + half * 16);
    v8h a1 = *(const v8h*)(srowp + kb + half * 16 + 8);
    v16h a = pack16(a0, a1);
#pragma unroll
    for (int nt = 0; nt < 4; nt++) {
      const _Float16* bp = t1xT + ((size_t)b * DIN_ + nt * 16 + lm) * NP + kb + half * 16;
      v16h bop = pack16(*(const v8h*)bp, *(const v8h*)(bp + 8));
      acc[nt] = __builtin_amdgcn_wmma_f32_16x16x32_f16(false, a, false, bop,
                                                       (short)0, acc[nt], false, false);
    }
  }
#pragma unroll
  for (int nt = 0; nt < 4; nt++) {
    int i = nt * 16 + lm;
#pragma unroll
    for (int j = 0; j < 8; j++) {
      int node = mt * 16 + j + half * 8;
      float v = 2.0f * acc[nt][j] - x[((size_t)b * NN + node) * DIN_ + i];
      xg[((size_t)b * NN + node) * KI_ + 128 + i] = (_Float16)v;  // k = 2 slot
    }
  }
}

// ---------------------------------------------------------------------------
// out[b,n,o] = sum_f fe[b,n,f] * ( (x_g[b,n,:] @ Wp[f,:,o]) + bias[f,o] )
// A-operands (x_g chunks) hoisted into registers; 16 x 24 WMMAs per wave.
// ---------------------------------------------------------------------------
__global__ void __launch_bounds__(32) k_out(const _Float16* __restrict__ xg,
                                            const _Float16* __restrict__ wpt,
                                            const float* __restrict__ fe32,
                                            const float* __restrict__ bias,
                                            float* __restrict__ out) {
  int blk = blockIdx.x;
  int b = blk / MT, mt = blk % MT;
  int L = threadIdx.x, lm = L & 15, half = L >> 4;
  int row = mt * 16 + lm;
  const _Float16* ap = xg + ((size_t)b * NN + row) * KI_;

  v16h A[6];
#pragma unroll
  for (int kc = 0; kc < 6; kc++) {
    v8h a0 = *(const v8h*)(ap + kc * 32 + half * 16);
    v8h a1 = *(const v8h*)(ap + kc * 32 + half * 16 + 8);
    A[kc] = pack16(a0, a1);
  }

  float outv[4][8];
#pragma unroll
  for (int nt = 0; nt < 4; nt++)
#pragma unroll
    for (int j = 0; j < 8; j++) outv[nt][j] = 0.0f;

  for (int f = 0; f < DF_; f++) {
    v8f g[4] = {};
#pragma unroll
    for (int kc = 0; kc < 6; kc++) {
#pragma unroll
      for (int nt = 0; nt < 4; nt++) {
        const _Float16* bp = wpt + ((size_t)f * DOUT_ + nt * 16 + lm) * KI_
                                 + kc * 32 + half * 16;
        v16h bop = pack16(*(const v8h*)bp, *(const v8h*)(bp + 8));
        g[nt] = __builtin_amdgcn_wmma_f32_16x16x32_f16(false, A[kc], false, bop,
                                                       (short)0, g[nt], false, false);
      }
    }
    float fev[8];
#pragma unroll
    for (int j = 0; j < 8; j++) {
      int node = mt * 16 + j + half * 8;
      fev[j] = fe32[((size_t)b * NN + node) * DF_ + f];
    }
#pragma unroll
    for (int nt = 0; nt < 4; nt++) {
      float bv = bias[f * DOUT_ + nt * 16 + lm];
#pragma unroll
      for (int j = 0; j < 8; j++) outv[nt][j] += fev[j] * (g[nt][j] + bv);
    }
  }

#pragma unroll
  for (int nt = 0; nt < 4; nt++) {
    int o = nt * 16 + lm;
#pragma unroll
    for (int j = 0; j < 8; j++) {
      int node = mt * 16 + j + half * 8;
      out[((size_t)b * NN + node) * DOUT_ + o] = outv[nt][j];
    }
  }
}

// ---------------------------------------------------------------------------
extern "C" void kernel_launch(void* const* d_in, const int* in_sizes, int n_in,
                              void* d_out, int out_size, void* d_ws, size_t ws_size,
                              hipStream_t stream) {
  const float* x    = (const float*)d_in[0];
  const float* pre  = (const float*)d_in[1];
  const float* unk  = (const float*)d_in[2];
  const float* Wfe  = (const float*)d_in[3];
  const float* bfe  = (const float*)d_in[4];
  const float* wp   = (const float*)d_in[5];
  const float* bias = (const float*)d_in[6];
  float* out = (float*)d_out;

  char* ws = (char*)d_ws;
  float*    fe32 = (float*)   (ws + OFF_FE32);
  _Float16* fe16 = (_Float16*)(ws + OFF_FE16);
  _Float16* sup  = (_Float16*)(ws + OFF_SUP);
  _Float16* xT   = (_Float16*)(ws + OFF_XT);
  _Float16* t1xT = (_Float16*)(ws + OFF_T1XT);
  _Float16* xg   = (_Float16*)(ws + OFF_XG);
  _Float16* wpt  = (_Float16*)(ws + OFF_WPT);

  (void)in_sizes; (void)n_in; (void)out_size; (void)ws_size;

  k_fe<<<(BB * NN * DF_ + 255) / 256, 256, 0, stream>>>(pre, unk, Wfe, bfe, fe32, fe16);
  k_xprep<<<(BB * DIN_ * NP + 255) / 256, 256, 0, stream>>>(x, xT, t1xT, xg);
  k_wpt<<<(DF_ * DOUT_ * KI_ + 255) / 256, 256, 0, stream>>>(wp, wpt);

  k_adj<<<BB * MT, 32, 0, stream>>>(fe16, sup);
  k_t1x<<<BB * MT, 32, 0, stream>>>(sup, xT, xg, t1xT);
  k_t2x<<<BB * MT, 32, 0, stream>>>(sup, t1xT, x, xg);
  k_out<<<BB * MT, 32, 0, stream>>>(xg, wpt, fe32, bias, out);
}
